// SparsemaxBertSelfAttention_59803124630067
// MI455X (gfx1250) — matrix operand order repaired and checked by reading, not compiled, not run on hardware
//
#include <hip/hip_runtime.h>
#include <hip/hip_bf16.h>

// ---------------------------------------------------------------------------
// SparsemaxBertSelfAttention for MI455X (gfx1250, wave32, WMMA)
// B=4, S=1024, H=16, HD=64, D=1024
// GEMMs on v_wmma_f32_16x16x32_bf16 with 2x4 register blocking (proj) to
// raise arithmetic intensity from 8 to ~21 FLOP/B (L2-traffic bound otherwise).
// Sparsemax tau via 30-step bisection; softmax on odd heads.
// ---------------------------------------------------------------------------

#define S_LEN 1024
#define D_DIM 1024
#define NHEAD 16
#define HDIM  64
#define BATCH 4

typedef __attribute__((ext_vector_type(16))) __bf16 v16bf;
typedef __attribute__((ext_vector_type(8)))  float  v8f;

union FragU { uint4 q[2]; v16bf v; };

__device__ __forceinline__ unsigned short f32_bf16(float f) {
    unsigned int u = __float_as_uint(f);
    u += 0x7FFFu + ((u >> 16) & 1u);          // round-to-nearest-even
    return (unsigned short)(u >> 16);
}

// A-fragment (16x32, M x K), src row-major [row][k], ld in halfwords.
__device__ __forceinline__ v16bf load_a_frag(const unsigned short* base, int ld,
                                             int row0, int k0) {
    int lane = threadIdx.x & 31;
    int r    = lane & 15;
    int koff = (lane >> 4) << 3;
    const unsigned short* p = base + (size_t)(row0 + r) * ld + k0 + koff;
    FragU f;
    f.q[0] = *(const uint4*)(p);
    f.q[1] = *(const uint4*)(p + 16);
    return f.v;
}

// B-fragment (32x16, K x N) from a K-contiguous ("transposed") source.
__device__ __forceinline__ v16bf load_b_frag(const unsigned short* baseT, int ld,
                                             int col0, int k0) {
    int lane = threadIdx.x & 31;
    int n    = lane & 15;
    int koff = (lane >> 4) << 3;
    const unsigned short* p = baseT + (size_t)(col0 + n) * ld + k0 + koff;
    FragU f;
    f.q[0] = *(const uint4*)(p);
    f.q[1] = *(const uint4*)(p + 16);
    return f.v;
}

#define WMMA_BF16(A, B, C) \
    __builtin_amdgcn_wmma_f32_16x16x32_bf16(false, (A), false, (B), (short)0, (C), false, false)

// ---------------------------------------------------------------------------
// Kernel 1a: fp32 -> bf16 hidden states (float4 / ushort4 vectorized)
// ---------------------------------------------------------------------------
__global__ void __launch_bounds__(256) hconv_kernel(
    const float* __restrict__ hs, unsigned short* __restrict__ hsb)
{
    const size_t total4 = (size_t)BATCH * S_LEN * D_DIM / 4;    // 1M
    for (size_t i = (size_t)blockIdx.x * 256 + threadIdx.x; i < total4;
         i += (size_t)gridDim.x * 256) {
        float4 f = ((const float4*)hs)[i];
        ushort4 o;
        o.x = f32_bf16(f.x); o.y = f32_bf16(f.y);
        o.z = f32_bf16(f.z); o.w = f32_bf16(f.w);
        ((ushort4*)hsb)[i] = o;
    }
}

// ---------------------------------------------------------------------------
// Kernel 1b: weight transpose+convert, coalesced both sides via 32x33 LDS tile.
// Wt[n][k] = bf16(W[k][n]).  grid (32, 32, 3), block 256.
// ---------------------------------------------------------------------------
__global__ void __launch_bounds__(256) wtrans_kernel(
    const float* __restrict__ Wq, const float* __restrict__ Wk,
    const float* __restrict__ Wv, unsigned short* __restrict__ wt)
{
    __shared__ float tile[32][33];
    const int mat = blockIdx.z;
    const float* W = (mat == 0) ? Wq : (mat == 1) ? Wk : Wv;
    const int k0 = blockIdx.y * 32;
    const int n0 = blockIdx.x * 32;
    const int c     = threadIdx.x & 31;
    const int rbase = (threadIdx.x >> 5) * 4;
#pragma unroll
    for (int i = 0; i < 4; ++i)
        tile[rbase + i][c] = W[(size_t)(k0 + rbase + i) * D_DIM + n0 + c];
    __syncthreads();
    unsigned short* dst = wt + (size_t)mat * D_DIM * D_DIM;
#pragma unroll
    for (int i = 0; i < 4; ++i)
        dst[(size_t)(n0 + rbase + i) * D_DIM + k0 + c] = f32_bf16(tile[c][rbase + i]);
}

// ---------------------------------------------------------------------------
// Kernel 2: QKV projection GEMM with 2x4 register blocking per wave.
// Wave computes rows [mt*32, mt*32+32) x cols [blockIdx.x*64, +64).
// Q scaled by HD^-0.5 -> [B,H,S,HD]; K -> [B,H,S,HD]; V -> transposed [B,H,HD,S].
// grid = (16, 16, 3), block = 256 (8 waves).
// ---------------------------------------------------------------------------
__global__ void __launch_bounds__(256) proj_kernel(
    const unsigned short* __restrict__ hsb,
    const unsigned short* __restrict__ wt,
    const float* __restrict__ bq, const float* __restrict__ bk,
    const float* __restrict__ bv,
    unsigned short* __restrict__ qb,
    unsigned short* __restrict__ kb,
    unsigned short* __restrict__ vt)
{
    const int mat  = blockIdx.z;
    const int wave = threadIdx.x >> 5;
    const int lane = threadIdx.x & 31;
    const int row0 = (blockIdx.y * 8 + wave) * 32;   // 2 M-tiles
    const int col0 = blockIdx.x * 64;                // 4 N-tiles
    const unsigned short* w = wt + (size_t)mat * D_DIM * D_DIM;
    const float* bias = (mat == 0) ? bq : (mat == 1) ? bk : bv;

    v8f acc[2][4] = {};
    for (int k0 = 0; k0 < D_DIM; k0 += 32) {
        v16bf a0 = load_a_frag(hsb, D_DIM, row0,      k0);
        v16bf a1 = load_a_frag(hsb, D_DIM, row0 + 16, k0);
        v16bf b0 = load_b_frag(w, D_DIM, col0,      k0);
        v16bf b1 = load_b_frag(w, D_DIM, col0 + 16, k0);
        v16bf b2 = load_b_frag(w, D_DIM, col0 + 32, k0);
        v16bf b3 = load_b_frag(w, D_DIM, col0 + 48, k0);
        acc[0][0] = WMMA_BF16(a0, b0, acc[0][0]);
        acc[0][1] = WMMA_BF16(a0, b1, acc[0][1]);
        acc[0][2] = WMMA_BF16(a0, b2, acc[0][2]);
        acc[0][3] = WMMA_BF16(a0, b3, acc[0][3]);
        acc[1][0] = WMMA_BF16(a1, b0, acc[1][0]);
        acc[1][1] = WMMA_BF16(a1, b1, acc[1][1]);
        acc[1][2] = WMMA_BF16(a1, b2, acc[1][2]);
        acc[1][3] = WMMA_BF16(a1, b3, acc[1][3]);
    }

    const int   n     = lane & 15;
    const int   mb    = (lane >> 4) * 8;      // C layout: vgpr i -> M = mb+i
    const float scale = (mat == 0) ? 0.125f : 1.0f;    // HD^-0.5 = 1/8
#pragma unroll
    for (int ni = 0; ni < 4; ++ni) {
        const int   col  = col0 + ni * 16 + n;
        const float bval = bias[col];
        const int   h    = col >> 6, hd = col & 63;
#pragma unroll
        for (int mi = 0; mi < 2; ++mi) {
#pragma unroll
            for (int i = 0; i < 8; ++i) {
                int row  = row0 + mi * 16 + mb + i;
                int bidx = row >> 10, s = row & 1023;
                unsigned short hv = f32_bf16((acc[mi][ni][i] + bval) * scale);
                if (mat == 0)
                    qb[(((size_t)(bidx * NHEAD + h)) * S_LEN + s) * HDIM + hd] = hv;
                else if (mat == 1)
                    kb[(((size_t)(bidx * NHEAD + h)) * S_LEN + s) * HDIM + hd] = hv;
                else
                    vt[(((size_t)(bidx * NHEAD + h)) * HDIM + hd) * S_LEN + s] = hv;
            }
        }
    }
}

// ---------------------------------------------------------------------------
// Kernel 3: attention.  One workgroup (8 waves) per (b, h, 16-query tile).
// ---------------------------------------------------------------------------
__global__ void __launch_bounds__(256) attn_kernel(
    const unsigned short* __restrict__ qb,
    const unsigned short* __restrict__ kb,
    const unsigned short* __restrict__ vt,
    const float* __restrict__ mask,
    float* __restrict__ out)
{
    __shared__ float sc[16][S_LEN];                          // 64 KB
    __shared__ alignas(16) unsigned short wb[16][S_LEN];     // 32 KB
    __shared__ float ot[16][HDIM];                           //  4 KB

    const int qt   = blockIdx.x;
    const int h    = blockIdx.y;
    const int b    = blockIdx.z;
    const int wave = threadIdx.x >> 5;
    const int lane = threadIdx.x & 31;
    const int q0   = qt * 16;

    const unsigned short* Q = qb + ((size_t)(b * NHEAD + h)) * S_LEN * HDIM;
    const unsigned short* K = kb + ((size_t)(b * NHEAD + h)) * S_LEN * HDIM;
    const unsigned short* V = vt + ((size_t)(b * NHEAD + h)) * HDIM * S_LEN;

    // zero the output accumulation tile (used in stage 3; barriers below cover it)
    for (int t = threadIdx.x; t < 16 * HDIM; t += 256)
        (&ot[0][0])[t] = 0.0f;

    // ---- stage 1: scores -> LDS.  Q fragments hoisted (invariant per wave) ----
    const v16bf qa0 = load_a_frag(Q, HDIM, q0, 0);
    const v16bf qa1 = load_a_frag(Q, HDIM, q0, 32);
    for (int kt = wave; kt < S_LEN / 16; kt += 8) {
        v8f acc = {};
        v16bf b0 = load_b_frag(K, HDIM, kt * 16, 0);
        v16bf b1 = load_b_frag(K, HDIM, kt * 16, 32);
        acc = WMMA_BF16(qa0, b0, acc);
        acc = WMMA_BF16(qa1, b1, acc);
        const int n   = lane & 15;
        const int mb  = (lane >> 4) * 8;
        const int key = kt * 16 + n;
        const float mv = mask[(size_t)b * S_LEN + key];
#pragma unroll
        for (int i = 0; i < 8; ++i)
            sc[mb + i][key] = fmaxf(acc[i] + mv, -10000.0f);
    }
    __syncthreads();

    // ---- stage 2: per-row sparsemax (even h) / softmax (odd h) ----
    for (int rr = 0; rr < 2; ++rr) {
        const int r = wave * 2 + rr;
        float z[32];
        float m = -3.4e38f;
#pragma unroll
        for (int j = 0; j < 32; ++j) {
            z[j] = sc[r][lane + j * 32];
            m = fmaxf(m, z[j]);
        }
#pragma unroll
        for (int o = 16; o > 0; o >>= 1) m = fmaxf(m, __shfl_xor(m, o, 32));

        if (h & 1) {                       // softmax
            float s = 0.f;
#pragma unroll
            for (int j = 0; j < 32; ++j) { z[j] = __expf(z[j] - m); s += z[j]; }
#pragma unroll
            for (int o = 16; o > 0; o >>= 1) s += __shfl_xor(s, o, 32);
            const float inv = 1.0f / s;
#pragma unroll
            for (int j = 0; j < 32; ++j)
                wb[r][lane + j * 32] = f32_bf16(z[j] * inv);
        } else {                           // sparsemax: bisect sum(max(z-t,0))=1
            float lo = m - 1.0f, hi = m;
            for (int it = 0; it < 30; ++it) {
                const float mid = 0.5f * (lo + hi);
                float s = 0.f;
#pragma unroll
                for (int j = 0; j < 32; ++j) s += fmaxf(z[j] - mid, 0.f);
#pragma unroll
                for (int o = 16; o > 0; o >>= 1) s += __shfl_xor(s, o, 32);
                if (s > 1.0f) lo = mid; else hi = mid;
            }
            const float tau = 0.5f * (lo + hi);
#pragma unroll
            for (int j = 0; j < 32; ++j)
                wb[r][lane + j * 32] = f32_bf16(fmaxf(z[j] - tau, 0.f));
        }
    }
    __syncthreads();

    // ---- stage 3: out = W · V, split-K over 8 waves, ds_add_f32 combine ----
    {
        const int dt = wave & 3;            // d-tile (HD/16 = 4)
        const int kh = wave >> 2;           // K half
        v8f acc = {};
        const int kbeg = kh * (S_LEN / 2);
        for (int k0 = kbeg; k0 < kbeg + S_LEN / 2; k0 += 32) {
            v16bf a  = load_a_frag(&wb[0][0], S_LEN, 0, k0);
            v16bf bf = load_b_frag(V, S_LEN, dt * 16, k0);
            acc = WMMA_BF16(a, bf, acc);
        }
        const int n  = lane & 15;
        const int mb = (lane >> 4) * 8;
        const int d  = dt * 16 + n;
#pragma unroll
        for (int i = 0; i < 8; ++i)
            atomicAdd(&ot[mb + i][d], acc[i]);
    }
    __syncthreads();

    // ---- coalesced writeback ----
    for (int t = threadIdx.x; t < 16 * HDIM; t += 256) {
        const int r = t >> 6, d = t & 63;
        out[(((size_t)b) * S_LEN + q0 + r) * D_DIM + h * HDIM + d] = ot[r][d];
    }
}

// ---------------------------------------------------------------------------
extern "C" void kernel_launch(void* const* d_in, const int* in_sizes, int n_in,
                              void* d_out, int out_size, void* d_ws, size_t ws_size,
                              hipStream_t stream) {
    const float* hs   = (const float*)d_in[0];
    const float* Wq   = (const float*)d_in[1];
    const float* bq   = (const float*)d_in[2];
    const float* Wk   = (const float*)d_in[3];
    const float* bk   = (const float*)d_in[4];
    const float* Wv   = (const float*)d_in[5];
    const float* bv   = (const float*)d_in[6];
    const float* mask = (const float*)d_in[7];

    char* ws = (char*)d_ws;
    const size_t MB = 1024 * 1024;
    unsigned short* hsb = (unsigned short*)(ws);             //  8 MB
    unsigned short* wt  = (unsigned short*)(ws + 8  * MB);   //  6 MB
    unsigned short* qb  = (unsigned short*)(ws + 14 * MB);   //  8 MB
    unsigned short* kb  = (unsigned short*)(ws + 22 * MB);   //  8 MB
    unsigned short* vt  = (unsigned short*)(ws + 30 * MB);   //  8 MB (ends 38MB)

    hconv_kernel<<<1024, 256, 0, stream>>>(hs, hsb);
    wtrans_kernel<<<dim3(32, 32, 3), 256, 0, stream>>>(Wq, Wk, Wv, wt);
    proj_kernel<<<dim3(16, 16, 3), 256, 0, stream>>>(hsb, wt, bq, bk, bv,
                                                     qb, kb, vt);
    attn_kernel<<<dim3(64, NHEAD, BATCH), 256, 0, stream>>>(qb, kb, vt, mask,
                                                            (float*)d_out);
}
